// VisionTransformer_21139829031382
// MI455X (gfx1250) — matrix operand order repaired and checked by reading, not compiled, not run on hardware
//
#include <hip/hip_runtime.h>
#include <math.h>

// ---------------------------------------------------------------------------
// ViT front-end fused kernel for MI455X (gfx1250, wave32, WMMA f16 16x16x32).
// One block = one batch image. 256 threads = 8 waves.
// ---------------------------------------------------------------------------

typedef __attribute__((ext_vector_type(8)))  _Float16 v8h;
typedef __attribute__((ext_vector_type(16))) _Float16 v16h;
typedef __attribute__((ext_vector_type(8)))  float    v8f;

#define NTHREADS 256
#define NWAVES   8

#define EMB    512
#define NTOK   65
#define MPAD   80      // tokens padded to 5*16
#define NP     64      // patches
#define INFEAT 48

// LDS leading dims (in halfs unless noted). Chosen so 16 consecutive rows hit
// distinct LDS banks for WMMA fragment loads (row stride bytes mod 256 != 0).
#define LD_TOK 520
#define LD_Q   520
#define LD_QT  72
#define LD_SC  88      // floats
#define LD_W   72
#define LD_P   72

// LDS layout (bytes): tok | q | qT | scores(f32) | w16 ; patches aliases scores
#define SM_TOK_OFF 0
#define SM_Q_OFF   (MPAD * LD_TOK)                   // halfs
#define SM_QT_OFF  (SM_Q_OFF + MPAD * LD_Q)          // halfs
#define SM_SC_OFFB ((SM_QT_OFF + EMB * LD_QT) * 2)   // bytes
#define SM_W_OFFB  (SM_SC_OFFB + MPAD * LD_SC * 4)
#define SM_BYTES   (SM_W_OFFB + MPAD * LD_W * 2)     // = 279808 B < 320 KB

// ---------------------------------------------------------------------------
// Fragment loaders (ISA 7.12.2 layouts, wave32).
// A (16x32 f16): lane l holds row M = l&15; halfs 0..7  = K k0+(l>>4)*8 .. +7
//                                          halfs 8..15 = K k0+16+(l>>4)*8 .. +7
// B (32x16 f16): lane l holds col N = l&15; same K order as A (halfs 0..7 =
//                k0+(l>>4)*8.., 8..15 = k0+16+(l>>4)*8..). B column = W row.
// C (16x16 f32): lane l, elem v -> row = v + (l>>4)*8, col = l&15.
// ---------------------------------------------------------------------------
__device__ __forceinline__ v16h frag_cat(v8h lo, v8h hi) {
  return __builtin_shufflevector(lo, hi, 0, 1, 2, 3, 4, 5, 6, 7,
                                         8, 9, 10, 11, 12, 13, 14, 15);
}

__device__ __forceinline__ v16h load_a(const _Float16* __restrict__ S, int ld,
                                       int row0, int k0, int lane) {
  const int r  = row0 + (lane & 15);
  const int hs = lane >> 4;
  const _Float16* p = S + r * ld + k0 + hs * 8;
  return frag_cat(*(const v8h*)p, *(const v8h*)(p + 16));
}

__device__ __forceinline__ v16h load_b(const _Float16* __restrict__ W, int ld,
                                       int n0, int k0, int lane) {
  const int n  = n0 + (lane & 15);
  const int hs = lane >> 4;
  const _Float16* p = W + n * ld + k0 + hs * 8;
  return frag_cat(*(const v8h*)p, *(const v8h*)(p + 16));
}

__device__ __forceinline__ v8f wmma_f16(v16h a, v16h b, v8f c) {
  return __builtin_amdgcn_wmma_f32_16x16x32_f16(false, a, false, b,
                                                (short)0, c, false, false);
}

// ---------------------------------------------------------------------------
// Kernel 1: convert weights to f16 once per call into d_ws.
// Wq16: [512][512], Wp16: [512][64] zero-padded (cols 48..63).
// ---------------------------------------------------------------------------
__global__ __launch_bounds__(NTHREADS)
void cvt_weights(const float* __restrict__ Wq, const float* __restrict__ Wp,
                 _Float16* __restrict__ Wq16, _Float16* __restrict__ Wp16) {
  const int i = blockIdx.x * blockDim.x + threadIdx.x;
  if (i < EMB * EMB) Wq16[i] = (_Float16)Wq[i];
  if (i < EMB * 64) {
    const int r = i >> 6, c = i & 63;
    Wp16[i] = (c < INFEAT) ? (_Float16)Wp[r * INFEAT + c] : (_Float16)0.0f;
  }
}

// ---------------------------------------------------------------------------
// Kernel 2: fused ViT front-end per batch element.
// ---------------------------------------------------------------------------
__global__ __launch_bounds__(NTHREADS)
void vit_fused(const float* __restrict__ x, const float* __restrict__ bp,
               const float* __restrict__ cls, const float* __restrict__ pos,
               const float* __restrict__ bq, const _Float16* __restrict__ Wq16,
               const _Float16* __restrict__ Wp16, float* __restrict__ out) {
  extern __shared__ char smem[];
  _Float16* tok16  = (_Float16*)smem + SM_TOK_OFF;
  _Float16* q16    = (_Float16*)smem + SM_Q_OFF;
  _Float16* qT16   = (_Float16*)smem + SM_QT_OFF;
  float*    scores = (float*)(smem + SM_SC_OFFB);
  _Float16* w16    = (_Float16*)(smem + SM_W_OFFB);
  _Float16* patch16 = (_Float16*)(smem + SM_SC_OFFB);   // alias (phases 0-1)

  const int tid  = threadIdx.x;
  const int lane = tid & 31;
  const int wave = tid >> 5;
  const int b    = blockIdx.x;
  const float* xb = x + (size_t)b * 3 * 32 * 32;

  // ---- Phase 0: patchify -> f16 [64][64] (K padded); tok row 0; pad rows ---
  for (int i = tid; i < NP * 64; i += NTHREADS) {
    const int p = i >> 6, f = i & 63;
    _Float16 v = (_Float16)0.0f;
    if (f < INFEAT) {
      const int gy = p >> 3, gx = p & 7;
      const int c = f >> 4, kh = (f >> 2) & 3, kw = f & 3;
      v = (_Float16)xb[c * 1024 + (gy * 4 + kh) * 32 + (gx * 4 + kw)];
    }
    patch16[p * LD_P + f] = v;
  }
  for (int c = tid; c < EMB; c += NTHREADS)
    tok16[c] = (_Float16)(cls[c] + pos[c]);
  for (int i = tid; i < (MPAD - NTOK) * LD_TOK; i += NTHREADS)
    tok16[NTOK * LD_TOK + i] = (_Float16)0.0f;
  __syncthreads();

  // ---- Phase 1: emb = patches @ Wp^T + bp + pos -> tok rows 1..64 (f16) ----
  // 4 mTiles x 32 nTiles; wave owns 4 consecutive nTiles, reuses A frags.
  for (int mT = 0; mT < 4; ++mT) {
    v8f acc[4];
#pragma unroll
    for (int j = 0; j < 4; ++j) acc[j] = (v8f){0.f,0.f,0.f,0.f,0.f,0.f,0.f,0.f};
#pragma unroll
    for (int k0 = 0; k0 < 64; k0 += 32) {
      const v16h a = load_a(patch16, LD_P, mT * 16, k0, lane);
#pragma unroll
      for (int j = 0; j < 4; ++j) {
        const v16h bb = load_b(Wp16, 64, (wave * 4 + j) * 16, k0, lane);
        acc[j] = wmma_f16(a, bb, acc[j]);
      }
    }
#pragma unroll
    for (int j = 0; j < 4; ++j) {
      const int col = (wave * 4 + j) * 16 + (lane & 15);
      const float bpv = bp[col];
      const int rbase = mT * 16 + (lane >> 4) * 8;
#pragma unroll
      for (int v = 0; v < 8; ++v) {
        const int row = rbase + v;                     // emb row 0..63
        const float val = acc[j][v] + bpv + pos[(row + 1) * EMB + col];
        tok16[(row + 1) * LD_TOK + col] = (_Float16)val;
      }
    }
  }
  __syncthreads();

  // ---- Phase 2: q = tok @ Wq^T + bq ; also build qT for the final GEMM -----
  for (int mT = 0; mT < 5; ++mT) {
    v8f acc[4];
#pragma unroll
    for (int j = 0; j < 4; ++j) acc[j] = (v8f){0.f,0.f,0.f,0.f,0.f,0.f,0.f,0.f};
#pragma unroll 4
    for (int k0 = 0; k0 < EMB; k0 += 32) {
      const v16h a = load_a(tok16, LD_TOK, mT * 16, k0, lane);
#pragma unroll
      for (int j = 0; j < 4; ++j) {
        const v16h bb = load_b(Wq16, EMB, (wave * 4 + j) * 16, k0, lane);
        acc[j] = wmma_f16(a, bb, acc[j]);
      }
    }
#pragma unroll
    for (int j = 0; j < 4; ++j) {
      const int col = (wave * 4 + j) * 16 + (lane & 15);
      const float bqv = bq[col];
      const int rbase = mT * 16 + (lane >> 4) * 8;
#pragma unroll
      for (int v = 0; v < 8; ++v) {
        const int row = rbase + v;                     // 0..79
        const _Float16 h = (_Float16)(acc[j][v] + bqv);
        q16[row * LD_Q + col] = h;
        if (row < 64) qT16[col * LD_QT + row] = h;     // k = row, d = col
      }
    }
  }
  __syncthreads();

  // ---- Phase 3: scores = (q @ q^T) * 1/sqrt(512) -> f32 LDS [80][88] ------
  const float scale = 0.044194173824159216f;           // 1/sqrt(512)
  for (int t = wave; t < 25; t += NWAVES) {
    const int mT = t / 5, nT = t % 5;
    v8f acc = (v8f){0.f,0.f,0.f,0.f,0.f,0.f,0.f,0.f};
#pragma unroll 4
    for (int k0 = 0; k0 < EMB; k0 += 32) {
      const v16h a  = load_a(q16, LD_Q, mT * 16, k0, lane);
      const v16h bb = load_b(q16, LD_Q, nT * 16, k0, lane); // B[k][j]=q[j][k]
      acc = wmma_f16(a, bb, acc);
    }
    const int col = nT * 16 + (lane & 15);
    const int rbase = mT * 16 + (lane >> 4) * 8;
#pragma unroll
    for (int v = 0; v < 8; ++v)
      scores[(rbase + v) * LD_SC + col] = acc[v] * scale;
  }
  __syncthreads();

  // ---- Phase 4: zero-pad w16, then column softmax (over query axis) -------
  for (int i = tid; i < MPAD * LD_W; i += NTHREADS) w16[i] = (_Float16)0.0f;
  __syncthreads();
  if (tid < NTOK) {
    const int k = tid;                                  // column
    float m = -INFINITY;
    for (int qi = 0; qi < NTOK; ++qi)
      m = fmaxf(m, scores[qi * LD_SC + k]);
    float s = 0.f;
    for (int qi = 0; qi < NTOK; ++qi)
      s += __expf(scores[qi * LD_SC + k] - m);
    const float inv = 1.f / s;
    for (int qi = 0; qi < NTOK; ++qi) {
      const float wv = __expf(scores[qi * LD_SC + k] - m) * inv;
      scores[qi * LD_SC + k] = wv;                      // f32 copy (k=64 path)
      w16[qi * LD_W + k] = (_Float16)wv;                // f16 for WMMA (k<64)
    }
  }
  __syncthreads();

  // ---- Phase 5: attention = weights @ q.  K=64 via WMMA + rank-1 (k=64) ---
  for (int mT = 0; mT < 5; ++mT) {
    v8f acc[4];
#pragma unroll
    for (int j = 0; j < 4; ++j) acc[j] = (v8f){0.f,0.f,0.f,0.f,0.f,0.f,0.f,0.f};
#pragma unroll
    for (int k0 = 0; k0 < 64; k0 += 32) {
      const v16h a = load_a(w16, LD_W, mT * 16, k0, lane);
#pragma unroll
      for (int j = 0; j < 4; ++j) {
        const v16h bb = load_b(qT16, LD_QT, (wave * 4 + j) * 16, k0, lane);
        acc[j] = wmma_f16(a, bb, acc[j]);
      }
    }
#pragma unroll
    for (int j = 0; j < 4; ++j) {
      const int col = (wave * 4 + j) * 16 + (lane & 15);
      const float qv = (float)q16[64 * LD_Q + col];     // q[k=64][d]
      const int rbase = mT * 16 + (lane >> 4) * 8;
#pragma unroll
      for (int v = 0; v < 8; ++v) {
        const int row = rbase + v;
        const float val = acc[j][v] + scores[row * LD_SC + 64] * qv;
        if (row < NTOK)
          out[(size_t)b * NTOK * EMB + (size_t)row * EMB + col] = val;
      }
    }
  }
}

// ---------------------------------------------------------------------------
extern "C" void kernel_launch(void* const* d_in, const int* in_sizes, int n_in,
                              void* d_out, int out_size, void* d_ws,
                              size_t ws_size, hipStream_t stream) {
  const float* x   = (const float*)d_in[0];
  const float* Wp  = (const float*)d_in[1];
  const float* bp  = (const float*)d_in[2];
  const float* cls = (const float*)d_in[3];
  const float* pos = (const float*)d_in[4];
  const float* Wq  = (const float*)d_in[5];
  const float* bq  = (const float*)d_in[6];
  float* out = (float*)d_out;

  const int B = in_sizes[0] / (3 * 32 * 32);

  _Float16* Wq16 = (_Float16*)d_ws;            // 512*512 halfs
  _Float16* Wp16 = Wq16 + EMB * EMB;           // 512*64  halfs (576 KB total)

  cvt_weights<<<(EMB * EMB + NTHREADS - 1) / NTHREADS, NTHREADS, 0, stream>>>(
      Wq, Wp, Wq16, Wp16);

  static int smem_raised = 0;  // host-side attribute set (idempotent, cheap)
  (void)smem_raised;
  hipFuncSetAttribute(reinterpret_cast<const void*>(vit_fused),
                      hipFuncAttributeMaxDynamicSharedMemorySize, SM_BYTES);

  vit_fused<<<B, NTHREADS, SM_BYTES, stream>>>(x, bp, cls, pos, bq, Wq16, Wp16,
                                               out);
}
